// Estor_raw_45595372814583
// MI455X (gfx1250) — matrix operand order repaired and checked by reading, not compiled, not run on hardware
//
#include <hip/hip_runtime.h>
#include <hip/hip_bf16.h>

// ---------------------------------------------------------------------------
// Problem constants (match reference)
// ---------------------------------------------------------------------------
#define BB 64
#define LL 512
#define HH 768
#define II 3072
#define TT 50
#define NLBL 9
#define SS 4096
#define RATE 0.5f
#define EPSF 1e-12f
#define ROWS (BB * LL)   // 32768
#define LDA 40           // padded LDS row stride (elements) -> 80B, kills bank-stride conflicts

typedef __bf16 bf16_t;
typedef __attribute__((ext_vector_type(16))) bf16_t v16bf;
typedef __attribute__((ext_vector_type(8)))  bf16_t v8bf;
typedef __attribute__((ext_vector_type(8)))  float  v8f;

union Frag16 { v16bf v; v8bf h[2]; };

static __device__ __forceinline__ bf16_t f2bf(float f) {
    unsigned u = __builtin_bit_cast(unsigned, f);
    unsigned r = (u + 0x7FFFu + ((u >> 16) & 1u)) >> 16;
    unsigned short s = (unsigned short)r;
    return __builtin_bit_cast(bf16_t, s);
}

// ---------------------------------------------------------------------------
// CDNA5 async global->LDS copy (16B per lane), tracked by ASYNCcnt.
// Inline asm (portable across ROCm7.2 / amdgpu-toolchain builtin arity skew).
// Generic LDS pointer low 32 bits == LDS byte offset (ISA §10.2 aperture rule).
// ---------------------------------------------------------------------------
static __device__ __forceinline__ void cp16_async(bf16_t* lds, const bf16_t* g) {
    unsigned ldsoff = (unsigned)(size_t)lds;
    asm volatile("global_load_async_to_lds_b128 %0, %1, off"
                 :: "v"(ldsoff), "v"(g) : "memory");
}

static __device__ __forceinline__ void wait_async_all() {
    asm volatile("s_wait_asynccnt 0x0" ::: "memory");
}

// stage one 128x32 bf16 tile (row-major, global row stride = gstride) into LDS
static __device__ __forceinline__ void stage_tile(bf16_t* lds, const bf16_t* __restrict__ g,
                                                  int gstride, int row0, int kk) {
    // 128 rows * 4 chunks of 16B = 512 chunks over 256 threads (2 each, no divergence)
#pragma unroll
    for (int c0 = 0; c0 < 2; ++c0) {
        const int c   = threadIdx.x + c0 * 256;
        const int r   = c >> 2;
        const int seg = c & 3;
        cp16_async(lds + r * LDA + seg * 8,
                   g + (size_t)(row0 + r) * gstride + kk + seg * 8);
    }
}

// ---------------------------------------------------------------------------
// 1) copy word_embedding -> x (fp32 workspace)
// ---------------------------------------------------------------------------
__global__ __launch_bounds__(256) void copy_kernel(const float4* __restrict__ src,
                                                   float4* __restrict__ dst, int n4) {
    for (int i = blockIdx.x * blockDim.x + threadIdx.x; i < n4; i += gridDim.x * blockDim.x)
        dst[i] = src[i];
}

// ---------------------------------------------------------------------------
// 2) span scatter: x[b, s:e, :] += RATE * tag_emb[tag]
// ---------------------------------------------------------------------------
__global__ __launch_bounds__(256) void scatter_kernel(float* __restrict__ x,
                                                      const float* __restrict__ tag_emb,
                                                      const int* __restrict__ span_b,
                                                      const int* __restrict__ span_tag,
                                                      const int* __restrict__ span_start,
                                                      const int* __restrict__ span_end) {
    const int s   = blockIdx.x;
    const int b   = span_b[s];
    const int tag = span_tag[s];
    const int st  = span_start[s];
    const int en  = span_end[s];
    const int total = (en - st) * HH;
    const float* te = tag_emb + (size_t)tag * HH;
    for (int idx = threadIdx.x; idx < total; idx += 256) {
        const int l = st + idx / HH;
        const int h = idx % HH;
        atomicAdd(&x[((size_t)b * LL + l) * HH + h], RATE * te[h]);
    }
}

// ---------------------------------------------------------------------------
// shared-memory block reduction helper
// ---------------------------------------------------------------------------
static __device__ __forceinline__ float block_reduce(float v, float* red) {
    const int tid = threadIdx.x;
    red[tid] = v;
    __syncthreads();
    for (int s = 128; s > 0; s >>= 1) {
        if (tid < s) red[tid] += red[tid + s];
        __syncthreads();
    }
    float r = red[0];
    __syncthreads();
    return r;
}

// ---------------------------------------------------------------------------
// 3) att LayerNorm in place; also emit bf16 copy for WMMA
// ---------------------------------------------------------------------------
__global__ __launch_bounds__(256) void ln_att_kernel(float* __restrict__ x,
                                                     bf16_t* __restrict__ xb,
                                                     const float* __restrict__ g,
                                                     const float* __restrict__ be) {
    __shared__ float red[256];
    const size_t base = (size_t)blockIdx.x * HH;
    float s = 0.f, ss = 0.f;
    for (int j = threadIdx.x; j < HH; j += 256) {
        float v = x[base + j];
        s += v; ss += v * v;
    }
    const float sum  = block_reduce(s,  red);
    const float sum2 = block_reduce(ss, red);
    const float mean = sum * (1.0f / HH);
    float var = sum2 * (1.0f / HH) - mean * mean;
    var = fmaxf(var, 0.f);
    const float rs = rsqrtf(var + EPSF);
    for (int j = threadIdx.x; j < HH; j += 256) {
        float v = (x[base + j] - mean) * rs * g[j] + be[j];
        x[base + j]  = v;
        xb[base + j] = f2bf(v);
    }
}

// ---------------------------------------------------------------------------
// 4) transpose fp32 [R][C] -> bf16 [C][R]
// ---------------------------------------------------------------------------
__global__ __launch_bounds__(256) void transpose_bf16_kernel(const float* __restrict__ src,
                                                             bf16_t* __restrict__ dst,
                                                             int R, int C) {
    const int t = blockIdx.x * blockDim.x + threadIdx.x;
    if (t >= R * C) return;
    const int c = t / R;
    const int r = t - c * R;
    dst[(size_t)c * R + r] = f2bf(src[(size_t)r * C + c]);
}

// ---------------------------------------------------------------------------
// WMMA GEMM, LDS double-buffered with async global->LDS DMA.
//   Block tile 128(M) x 128(N), K-step 32; 8 waves as 2(M) x 4(N);
//   each wave: 4 M-tiles x 2 N-tiles of v_wmma_f32_16x16x32_bf16.
//   A  : [M][KTOT] bf16 row-major
//   Bt : [N][KTOT] bf16 (B transposed -> B-fragment is two 16B runs along k)
//   MODE 0: out = relu(acc + bias) -> bf16   (Nout = II)
//   MODE 1: out = acc + bias + Xres -> fp32  (Nout = HH, safe in-place over Xres)
// ---------------------------------------------------------------------------
template <int KTOT, int MODE>
__global__ __launch_bounds__(256, 1) void gemm_kernel(const bf16_t* __restrict__ A,
                                                      const bf16_t* __restrict__ Bt,
                                                      const float* __restrict__ bias,
                                                      const float* __restrict__ Xres,
                                                      bf16_t* __restrict__ outb,
                                                      float* __restrict__ outf,
                                                      int Nout) {
    __shared__ bf16_t lA[2][128 * LDA];
    __shared__ bf16_t lB[2][128 * LDA];

    const int wave  = threadIdx.x >> 5;
    const int wm    = wave >> 2;          // 0..1
    const int wn    = wave & 3;           // 0..3
    const int lane  = threadIdx.x & 31;
    const int khalf = lane >> 4;
    const int lidx  = lane & 15;
    const int blockM0 = blockIdx.y * 128;
    const int blockN0 = blockIdx.x * 128;
    const int tileM0  = blockM0 + wm * 64;
    const int tileN0  = blockN0 + wn * 32;

    v8f acc[4][2];
#pragma unroll
    for (int i = 0; i < 4; ++i)
#pragma unroll
        for (int j = 0; j < 2; ++j) acc[i][j] = (v8f)0.f;

    // prologue: DMA first K-tile
    stage_tile(lA[0], A,  KTOT, blockM0, 0);
    stage_tile(lB[0], Bt, KTOT, blockN0, 0);
    wait_async_all();
    __syncthreads();

    int p = 0;
    for (int k0 = 0; k0 < KTOT; k0 += 32, p ^= 1) {
        if (k0 + 32 < KTOT) {           // kick off DMA of next tile into other buffer
            stage_tile(lA[p ^ 1], A,  KTOT, blockM0, k0 + 32);
            stage_tile(lB[p ^ 1], Bt, KTOT, blockN0, k0 + 32);
        }

        Frag16 afr[4];
#pragma unroll
        for (int i = 0; i < 4; ++i) {
            const bf16_t* ap = &lA[p][(wm * 64 + i * 16 + lidx) * LDA + khalf * 8];
            afr[i].h[0] = *reinterpret_cast<const v8bf*>(ap);
            afr[i].h[1] = *reinterpret_cast<const v8bf*>(ap + 16);
        }
        Frag16 bfr[2];
#pragma unroll
        for (int j = 0; j < 2; ++j) {
            const bf16_t* bp = &lB[p][(wn * 32 + j * 16 + lidx) * LDA + khalf * 8];
            bfr[j].h[0] = *reinterpret_cast<const v8bf*>(bp);
            bfr[j].h[1] = *reinterpret_cast<const v8bf*>(bp + 16);
        }

#pragma unroll
        for (int i = 0; i < 4; ++i)
#pragma unroll
            for (int j = 0; j < 2; ++j)
                acc[i][j] = __builtin_amdgcn_wmma_f32_16x16x32_bf16(
                    false, afr[i].v, false, bfr[j].v, (short)0, acc[i][j], false, false);

        wait_async_all();   // next tile fully in LDS (this wave's DMAs)
        __syncthreads();    // everyone done reading buffer p; next tile visible
    }

    // epilogue (C/D layout: lane -> n = lane&15, m = (lane>>4)*8 + r)
    const int nloc = lidx;
    const int mhi  = khalf * 8;
#pragma unroll
    for (int i = 0; i < 4; ++i)
#pragma unroll
        for (int j = 0; j < 2; ++j) {
            const int col = tileN0 + j * 16 + nloc;
            const float bv = bias[col];
#pragma unroll
            for (int r = 0; r < 8; ++r) {
                const int row = tileM0 + i * 16 + mhi + r;
                if (MODE == 0) {
                    float v = acc[i][j][r] + bv;
                    v = v > 0.f ? v : 0.f;
                    outb[(size_t)row * Nout + col] = f2bf(v);
                } else {
                    const size_t off = (size_t)row * Nout + col;
                    outf[off] = acc[i][j][r] + bv + Xres[off];
                }
            }
        }
}

// ---------------------------------------------------------------------------
// 7) ff LayerNorm + 768->9 projection (+ b_out). One block per row.
// ---------------------------------------------------------------------------
__global__ __launch_bounds__(256) void final_kernel(const float* __restrict__ z,
                                                    const float* __restrict__ g,
                                                    const float* __restrict__ be,
                                                    const float* __restrict__ wout,
                                                    const float* __restrict__ bout,
                                                    float* __restrict__ out) {
    __shared__ float red[256];
    const size_t base = (size_t)blockIdx.x * HH;
    float s = 0.f, ss = 0.f;
    for (int j = threadIdx.x; j < HH; j += 256) {
        float v = z[base + j];
        s += v; ss += v * v;
    }
    const float sum  = block_reduce(s,  red);
    const float sum2 = block_reduce(ss, red);
    const float mean = sum * (1.0f / HH);
    float var = sum2 * (1.0f / HH) - mean * mean;
    var = fmaxf(var, 0.f);
    const float rs = rsqrtf(var + EPSF);

    float o[NLBL];
#pragma unroll
    for (int n = 0; n < NLBL; ++n) o[n] = 0.f;
    for (int j = threadIdx.x; j < HH; j += 256) {
        const float h = (z[base + j] - mean) * rs * g[j] + be[j];
        const float* wr = wout + (size_t)j * NLBL;
#pragma unroll
        for (int n = 0; n < NLBL; ++n) o[n] += h * wr[n];
    }
#pragma unroll
    for (int n = 0; n < NLBL; ++n) {
        const float t = block_reduce(o[n], red);
        if (threadIdx.x == 0) out[(size_t)blockIdx.x * NLBL + n] = t + bout[n];
    }
}

// ---------------------------------------------------------------------------
// launch
// ---------------------------------------------------------------------------
extern "C" void kernel_launch(void* const* d_in, const int* in_sizes, int n_in,
                              void* d_out, int out_size, void* d_ws, size_t ws_size,
                              hipStream_t stream) {
    const float* word_embedding = (const float*)d_in[0];
    const float* tag_emb        = (const float*)d_in[1];
    const float* w1             = (const float*)d_in[2];
    const float* b1             = (const float*)d_in[3];
    const float* w2             = (const float*)d_in[4];
    const float* b2             = (const float*)d_in[5];
    const float* g_att  = (const float*)d_in[6];
    const float* be_att = (const float*)d_in[7];
    const float* g_ff   = (const float*)d_in[8];
    const float* be_ff  = (const float*)d_in[9];
    const float* w_out  = (const float*)d_in[10];
    const float* b_out  = (const float*)d_in[11];
    const int* span_b     = (const int*)d_in[12];
    const int* span_tag   = (const int*)d_in[13];
    const int* span_start = (const int*)d_in[14];
    const int* span_end   = (const int*)d_in[15];

    float* outp = (float*)d_out;

    // workspace carve-up (all offsets 256B aligned)
    char* ws = (char*)d_ws;
    float*  x   = (float*)(ws + 0);                           // 32768*768*4  = 100663296
    bf16_t* xb  = (bf16_t*)(ws + 100663296);                  // 32768*768*2  =  50331648
    bf16_t* y   = (bf16_t*)(ws + 150994944);                  // 32768*3072*2 = 201326592
    bf16_t* w1t = (bf16_t*)(ws + 352321536);                  // 3072*768*2   =   4718592
    bf16_t* w2t = (bf16_t*)(ws + 357040128);                  // 768*3072*2   =   4718592
    (void)ws_size; (void)in_sizes; (void)n_in; (void)out_size;

    // 1) x = word_embedding
    const int n4 = (ROWS * HH) / 4;
    copy_kernel<<<4096, 256, 0, stream>>>((const float4*)word_embedding, (float4*)x, n4);

    // 2) span scatter-add
    scatter_kernel<<<SS, 256, 0, stream>>>(x, tag_emb, span_b, span_tag, span_start, span_end);

    // 3) att layernorm (in place) + bf16 copy
    ln_att_kernel<<<ROWS, 256, 0, stream>>>(x, xb, g_att, be_att);

    // 4) weight transposes -> bf16 (B-operand column-contiguous)
    transpose_bf16_kernel<<<(HH * II + 255) / 256, 256, 0, stream>>>(w1, w1t, HH, II);
    transpose_bf16_kernel<<<(II * HH + 255) / 256, 256, 0, stream>>>(w2, w2t, II, HH);

    // 5) Y = relu(X @ W1 + b1)   [32768 x 3072]
    gemm_kernel<HH, 0><<<dim3(II / 128, ROWS / 128), 256, 0, stream>>>(
        xb, w1t, b1, nullptr, y, nullptr, II);

    // 6) Z = Y @ W2 + b2 + X  (in-place over x)
    gemm_kernel<II, 1><<<dim3(HH / 128, ROWS / 128), 256, 0, stream>>>(
        y, w2t, b2, x, nullptr, x, HH);

    // 7) ff layernorm + 768->9 projection
    final_kernel<<<ROWS, 256, 0, stream>>>(x, g_ff, be_ff, w_out, b_out, outp);
}
